// PNAnet4L_80264348827994
// MI455X (gfx1250) — matrix operand order repaired and checked by reading, not compiled, not run on hardware
//
#include <hip/hip_runtime.h>
#include <cmath>

#define NN 30000
#define NE 480000
#define FIN 64
#define HD 128
#define BD 10
#define EPSV 1e-5f
#define AVG_LOG 2.8332133440562162f  // log(17)

typedef __bf16 bf16;
typedef __attribute__((ext_vector_type(16))) __bf16 v16bf;
typedef __attribute__((ext_vector_type(8)))  __bf16 v8bf;
typedef __attribute__((ext_vector_type(4)))  __bf16 v4bf;
typedef __attribute__((ext_vector_type(8)))  float  v8f;

// ---------------------------------------------------------------------------
// W pre-pack: f32 [K,N] row-major -> bf16 in WMMA-B fragment order:
//   Wp[((t*2+half)*N + n)*16 + k16]  <- W[(t*32 + half*16 + k16)*N + n]
// i.e. lane (half, n)'s 16 contiguous K-values are 32 contiguous bytes.
// Note (t*2+half)*16 + k16 == k, so the mapping collapses nicely.
// ---------------------------------------------------------------------------
__global__ void pack_w_kernel(const float* __restrict__ W, bf16* __restrict__ Wp,
                              int K, int N) {
    long long i = (long long)blockIdx.x * blockDim.x + threadIdx.x;
    if (i >= (long long)K * N) return;
    int k16 = (int)(i & 15);
    long long r = i >> 4;
    int n = (int)(r % N);
    int th = (int)(r / N);            // t*2 + half
    int k = th * 16 + k16;
    Wp[i] = (bf16)W[(long long)k * N + n];
}

// ---------------------------------------------------------------------------
// WMMA GEMM: C[M,N] (=|+=) gather(A)[M,K] @ W[K,N] (+bias) (opt relu)
// One wave -> 16x16 tile; NW waves cover 16 rows x NW*16 cols.
// M % 16 == 0 and K % 32 == 0 guaranteed by the model shapes -> no guards.
// ---------------------------------------------------------------------------
template <int NW, bool GATHER>
__global__ __launch_bounds__(NW * 32)
void wmma_gemm_kernel(const float* __restrict__ A, const int* __restrict__ idx,
                      int K, int lda,
                      const bf16* __restrict__ Wp, const float* __restrict__ bias,
                      float* __restrict__ C, int ldc, int accum, int relu)
{
    constexpr int Nn = NW * 16;
    __shared__ bf16 As[16][40];       // 80B row pitch: b128-aligned, bank-conflict-free
    const int tid  = threadIdx.x;
    const int wave = tid >> 5;
    const int lane = tid & 31;
    const int half = lane >> 4;
    const int ml   = lane & 15;
    const int m0   = blockIdx.x * 16;
    const int n    = wave * 16 + ml;

    // staging: 128 units = 16 rows x 8 float4 chunks; gather base hoisted out of K loop
    const int sr = tid >> 3;
    const int sc = tid & 7;
    const float* arow = A;
    if (sr < 16) {
        int row = m0 + sr;
        int ar = GATHER ? idx[row] : row;
        arow = A + (size_t)ar * lda + sc * 4;
    }

    v8f acc;
    if (accum) {
#pragma unroll
        for (int r = 0; r < 8; ++r)
            acc[r] = C[(size_t)(m0 + r + 8 * half) * ldc + n];
    } else {
        float bv = bias ? bias[n] : 0.f;
#pragma unroll
        for (int r = 0; r < 8; ++r) acc[r] = bv;
    }

    union BV { v16bf v; v8bf h[2]; };

    for (int k0 = 0; k0 < K; k0 += 32) {
        if (sr < 16) {
            float4 fv = *(const float4*)(arow + k0);
            v4bf b4;
            b4[0] = (bf16)fv.x; b4[1] = (bf16)fv.y;
            b4[2] = (bf16)fv.z; b4[3] = (bf16)fv.w;
            *(v4bf*)&As[sr][sc * 4] = b4;
        }
        __syncthreads();

        BV aF, bF;
        // A fragment: lane = row ml; two contiguous 8-value K-runs -> 2x ds_load_b128
        aF.h[0] = *(const v8bf*)&As[ml][16 * half];
        aF.h[1] = *(const v8bf*)&As[ml][16 * half + 8];
        // B fragment: packed contiguous 32B per (ktile, half, n) -> 2x global b128
        const bf16* bp = Wp + ((size_t)((k0 >> 5) * 2 + half) * Nn + n) * 16;
        bF.h[0] = *(const v8bf*)bp;
        bF.h[1] = *(const v8bf*)(bp + 8);

        acc = __builtin_amdgcn_wmma_f32_16x16x32_bf16(false, aF.v, false, bF.v,
                                                      (short)0, acc, false, false);
        __syncthreads();
    }

#pragma unroll
    for (int r = 0; r < 8; ++r) {
        float v = acc[r];
        if (relu) v = fmaxf(v, 0.f);
        C[(size_t)(m0 + r + 8 * half) * ldc + n] = v;
    }
}

// ---------------------------------------------------------------------------
// scalar / atomic helpers
// ---------------------------------------------------------------------------
__device__ inline void atomicMaxF(float* a, float v) {
    if (v >= 0.f) atomicMax((int*)a, __float_as_int(v));
    else          atomicMin((unsigned int*)a, __float_as_uint(v));
}
__device__ inline void atomicMinF(float* a, float v) {
    if (v >= 0.f) atomicMin((int*)a, __float_as_int(v));
    else          atomicMax((unsigned int*)a, __float_as_uint(v));
}

__global__ void fill_kernel(float* p, long long n, float v) {
    long long i = (long long)blockIdx.x * blockDim.x + threadIdx.x;
    if (i < n) p[i] = v;
}

// Wf[4,F] = We(4,F) @ W3(F,F); bf[F] = preb + be @ W3  (edge-encoder folded thru W3)
__global__ void fuse_edge_kernel(const float* __restrict__ We, const float* __restrict__ be,
                                 const float* __restrict__ preW, const float* __restrict__ preb,
                                 int F, float* __restrict__ Wf, float* __restrict__ bf) {
    int i = blockIdx.x * blockDim.x + threadIdx.x;
    if (i >= 5 * F) return;
    int k = i / F, f = i % F;
    const float* W3 = preW + 2 * F * F;
    if (k < 4) {
        float acc = 0.f;
        for (int j = 0; j < F; ++j) acc += We[k * F + j] * W3[j * F + f];
        Wf[k * F + f] = acc;
    } else {
        float acc = preb[f];
        for (int j = 0; j < F; ++j) acc += be[j] * W3[j * F + f];
        bf[f] = acc;
    }
}

// h[e,f] = bf[f] + sum_k ea[e,k]*Wf[k,f]
__global__ void edge_init_kernel(const float* __restrict__ ea, const float* __restrict__ Wf,
                                 const float* __restrict__ bf, float* __restrict__ h,
                                 int E, int F) {
    long long i = (long long)blockIdx.x * blockDim.x + threadIdx.x;
    if (i >= (long long)E * F) return;
    int e = (int)(i / F), f = (int)(i % F);
    const float* a = ea + 4 * e;
    h[i] = bf[f] + a[0] * Wf[f] + a[1] * Wf[F + f] + a[2] * Wf[2 * F + f] + a[3] * Wf[3 * F + f];
}

__global__ void count_kernel(const int* __restrict__ tgt, float* __restrict__ cnt, int E) {
    int i = blockIdx.x * blockDim.x + threadIdx.x;
    if (i < E) atomicAdd(&cnt[tgt[i]], 1.f);
}

__global__ void aggregate_kernel(const float* __restrict__ h, const int* __restrict__ tgt,
                                 float* __restrict__ s, float* __restrict__ sq,
                                 float* __restrict__ mn, float* __restrict__ mx,
                                 int E, int F) {
    long long i = (long long)blockIdx.x * blockDim.x + threadIdx.x;
    if (i >= (long long)E * F) return;
    int e = (int)(i / F), f = (int)(i % F);
    float v = h[i];
    long long o = (long long)tgt[e] * F + f;
    atomicAdd(&s[o], v);
    atomicAdd(&sq[o], v * v);
    atomicMaxF(&mx[o], v);
    atomicMinF(&mn[o], v);
}

// P[n, 13F] = [x | mean mn mx std | *logd/AVG | *AVG/logd]
__global__ void build_post_kernel(const float* __restrict__ x,
                                  const float* __restrict__ s, const float* __restrict__ sq,
                                  const float* __restrict__ mn, const float* __restrict__ mx,
                                  const float* __restrict__ cnt, float* __restrict__ P,
                                  int Nn, int F) {
    long long i = (long long)blockIdx.x * blockDim.x + threadIdx.x;
    if (i >= (long long)Nn * F) return;
    int n = (int)(i / F), f = (int)(i % F);
    float c  = cnt[n];
    float c1 = fmaxf(c, 1.f);
    long long o = (long long)n * F + f;
    float mean = s[o] / c1;
    float stdv = sqrtf(fmaxf(sq[o] / c1 - mean * mean, 0.f) + EPSV);
    bool has = c > 0.f;
    float mnv = has ? mn[o] : 0.f;
    float mxv = has ? mx[o] : 0.f;
    float logd = logf(c1 + 1.f);
    float s1 = logd / AVG_LOG, s2 = AVG_LOG / logd;
    long long base = (long long)n * 13 * F;
    P[base + f] = x[o];
    float ag[4] = {mean, mnv, mxv, stdv};
#pragma unroll
    for (int a = 0; a < 4; ++a) {
        P[base + F     + a * F + f] = ag[a];
        P[base + 5 * F + a * F + f] = ag[a] * s1;
        P[base + 9 * F + a * F + f] = ag[a] * s2;
    }
}

__global__ void colstats_kernel(const float* __restrict__ X, int Nn, int F,
                                float* __restrict__ cs, float* __restrict__ cq) {
    long long i = (long long)blockIdx.x * blockDim.x + threadIdx.x;
    if (i >= (long long)Nn * F) return;
    int f = (int)(i % F);
    float v = X[i];
    atomicAdd(&cs[f], v);
    atomicAdd(&cq[f], v * v);
}

__global__ void bn_relu_kernel(const float* __restrict__ X, const float* __restrict__ cs,
                               const float* __restrict__ cq, const float* __restrict__ g,
                               const float* __restrict__ b, float* __restrict__ Y,
                               int Nn, int F) {
    long long i = (long long)blockIdx.x * blockDim.x + threadIdx.x;
    if (i >= (long long)Nn * F) return;
    int f = (int)(i % F);
    float m = cs[f] / (float)Nn;
    float var = cq[f] / (float)Nn - m * m;
    float y = (X[i] - m) * rsqrtf(var + EPSV) * g[f] + b[f];
    Y[i] = fmaxf(y, 0.f);
}

// ---------------------------------------------------------------------------
// Set2Set (1 step, zero initial state) + head MLPs
// ---------------------------------------------------------------------------
__global__ void s2s_q_kernel(const float* __restrict__ bih, const float* __restrict__ bhh,
                             float* __restrict__ q, int Bn, int Hn) {
    int i = blockIdx.x * blockDim.x + threadIdx.x;
    if (i >= Bn * Hn) return;
    int f = i % Hn;
    float ig = bih[f]          + bhh[f];
    float gg = bih[2 * Hn + f] + bhh[2 * Hn + f];
    float og = bih[3 * Hn + f] + bhh[3 * Hn + f];
    float cc = (1.f / (1.f + expf(-ig))) * tanhf(gg);   // f-gate * c0(=0) dropped
    q[i] = (1.f / (1.f + expf(-og))) * tanhf(cc);
}

__global__ void s2s_e_kernel(const float* __restrict__ X, const int* __restrict__ batch,
                             const float* __restrict__ q, float* __restrict__ ev,
                             int Nn, int Hn) {
    int n = blockIdx.x * blockDim.x + threadIdx.x;
    if (n >= Nn) return;
    const float* xr = X + (long)n * Hn;
    const float* qr = q + (long)batch[n] * Hn;
    float acc = 0.f;
    for (int f = 0; f < Hn; ++f) acc += xr[f] * qr[f];
    ev[n] = acc;
}

__global__ void s2s_max_kernel(const float* __restrict__ ev, const int* __restrict__ batch,
                               float* __restrict__ ms, int Nn) {
    int n = blockIdx.x * blockDim.x + threadIdx.x;
    if (n < Nn) atomicMaxF(&ms[batch[n]], ev[n]);
}

__global__ void s2s_a_kernel(const float* __restrict__ ev, const int* __restrict__ batch,
                             const float* __restrict__ ms, float* __restrict__ av,
                             float* __restrict__ asum, int Nn) {
    int n = blockIdx.x * blockDim.x + threadIdx.x;
    if (n >= Nn) return;
    int b = batch[n];
    float a = expf(ev[n] - ms[b]);
    av[n] = a;
    atomicAdd(&asum[b], a);
}

__global__ void s2s_r_kernel(const float* __restrict__ X, const int* __restrict__ batch,
                             const float* __restrict__ av, const float* __restrict__ asum,
                             float* __restrict__ rr, int Nn, int Hn) {
    long long i = (long long)blockIdx.x * blockDim.x + threadIdx.x;
    if (i >= (long long)Nn * Hn) return;
    int n = (int)(i / Hn), f = (int)(i % Hn);
    int b = batch[n];
    atomicAdd(&rr[b * Hn + f], av[n] / fmaxf(asum[b], 1e-20f) * X[i]);
}

__global__ void s2s_qstar_kernel(const float* __restrict__ q, const float* __restrict__ rr,
                                 float* __restrict__ qs, int Bn, int Hn) {
    int i = blockIdx.x * blockDim.x + threadIdx.x;
    if (i >= Bn * 2 * Hn) return;
    int b = i / (2 * Hn), f = i % (2 * Hn);
    qs[i] = (f < Hn) ? q[b * Hn + f] : rr[b * Hn + (f - Hn)];
}

__global__ void mlp_kernel(const float* __restrict__ X, int M, int K,
                           const float* __restrict__ W, const float* __restrict__ bias,
                           float* __restrict__ Y, int N, int relu) {
    int i = blockIdx.x * blockDim.x + threadIdx.x;
    if (i >= M * N) return;
    int m = i / N, n = i % N;
    float acc = bias[n];
    for (int k = 0; k < K; ++k) acc += X[m * K + k] * W[k * N + n];
    Y[i] = relu ? fmaxf(acc, 0.f) : acc;
}

// ---------------------------------------------------------------------------
static inline int cdiv(long long a, int b) { return (int)((a + b - 1) / b); }

static void gemm(const float* A, const int* idx, int M, int K, int lda,
                 const float* W, bf16* wpack, const float* bias, float* C,
                 int N, int ldc, int accum, int relu, hipStream_t s) {
    pack_w_kernel<<<cdiv((long long)K * N, 256), 256, 0, s>>>(W, wpack, K, N);
    dim3 grid(M / 16);
    if (N == 64) {
        if (idx) wmma_gemm_kernel<4, true ><<<grid, 128, 0, s>>>(A, idx, K, lda, wpack, bias, C, ldc, accum, relu);
        else     wmma_gemm_kernel<4, false><<<grid, 128, 0, s>>>(A, idx, K, lda, wpack, bias, C, ldc, accum, relu);
    } else {
        if (idx) wmma_gemm_kernel<8, true ><<<grid, 256, 0, s>>>(A, idx, K, lda, wpack, bias, C, ldc, accum, relu);
        else     wmma_gemm_kernel<8, false><<<grid, 256, 0, s>>>(A, idx, K, lda, wpack, bias, C, ldc, accum, relu);
    }
}

extern "C" void kernel_launch(void* const* d_in, const int* in_sizes, int n_in,
                              void* d_out, int out_size, void* d_ws, size_t ws_size,
                              hipStream_t stream) {
    const float* x0    = (const float*)d_in[0];
    const int*   eidx  = (const int*)  d_in[1];
    const int*   srcI  = eidx;
    const int*   tgtI  = eidx + NE;
    const float* eattr = (const float*)d_in[2];
    const int*   batch = (const int*)  d_in[4];

    struct ConvP { const float *bn_b,*bn_g,*edgeW,*edgeb,*linW,*linb,*postW,*postb,*preW,*preb; };
    ConvP cv[4];
    int pi = 8;
    for (int l = 0; l < 4; ++l) {
        cv[l].bn_b  = (const float*)d_in[pi++];
        cv[l].bn_g  = (const float*)d_in[pi++];
        cv[l].edgeW = (const float*)d_in[pi++];
        cv[l].edgeb = (const float*)d_in[pi++];
        cv[l].linW  = (const float*)d_in[pi++];
        cv[l].linb  = (const float*)d_in[pi++];
        cv[l].postW = (const float*)d_in[pi++];
        cv[l].postb = (const float*)d_in[pi++];
        cv[l].preW  = (const float*)d_in[pi++];
        cv[l].preb  = (const float*)d_in[pi++];
    }
    const float* lin1W = (const float*)d_in[48];
    const float* lin1b = (const float*)d_in[49];
    const float* lin2W = (const float*)d_in[50];
    const float* lin2b = (const float*)d_in[51];
    const float* lin3W = (const float*)d_in[52];
    const float* lin3b = (const float*)d_in[53];
    const float* bhh   = (const float*)d_in[56];
    const float* bih   = (const float*)d_in[57];

    // workspace carve-up (floats)
    float* ws = (float*)d_ws;
    size_t off = 0;
    auto alloc = [&](size_t n) { float* p = ws + off; off += n; return p; };
    float* h     = alloc((size_t)NE * HD);
    float* P     = alloc((size_t)NN * 13 * HD);
    float* sBuf  = alloc((size_t)NN * HD);   // reused as post-GEMM out
    float* sqBuf = alloc((size_t)NN * HD);   // reused as lin-GEMM out
    float* mnBuf = alloc((size_t)NN * HD);
    float* mxBuf = alloc((size_t)NN * HD);
    float* cnt   = alloc(NN);
    float* xA    = alloc((size_t)NN * HD);
    float* xB    = alloc((size_t)NN * HD);
    bf16*  wpack = (bf16*)alloc((size_t)13 * HD * HD / 2);  // max K*N = 1664*128 bf16
    float* csum  = alloc(HD);
    float* csq   = alloc(HD);
    float* wfuse = alloc(4 * HD);
    float* bfuse = alloc(HD);
    float* q     = alloc(BD * HD);
    float* ev    = alloc(NN);
    float* mseg  = alloc(BD);
    float* av    = alloc(NN);
    float* asum  = alloc(BD);
    float* rr    = alloc(BD * HD);
    float* qstar = alloc(BD * 2 * HD);
    float* t1    = alloc(BD * HD);
    float* t2    = alloc(BD * 64);
    if (off * sizeof(float) > ws_size) return;  // workspace too small: bail

    const float* xcur = x0;
    int F = FIN;
    float* xping = xA;
    float* xpong = xB;

    for (int l = 0; l < 4; ++l) {
        long long EF = (long long)NE * F;
        long long NF = (long long)NN * F;

        fuse_edge_kernel<<<cdiv(5 * F, 128), 128, 0, stream>>>(
            cv[l].edgeW, cv[l].edgeb, cv[l].preW, cv[l].preb, F, wfuse, bfuse);
        edge_init_kernel<<<cdiv(EF, 256), 256, 0, stream>>>(eattr, wfuse, bfuse, h, NE, F);

        // h += x[tgt]@W1 ; h += x[src]@W2   (WMMA, bf16 inputs, f32 accum)
        gemm(xcur, tgtI, NE, F, F, cv[l].preW,         wpack, nullptr, h, F, F, 1, 0, stream);
        gemm(xcur, srcI, NE, F, F, cv[l].preW + F * F, wpack, nullptr, h, F, F, 1, 0, stream);

        fill_kernel<<<cdiv(NF, 256), 256, 0, stream>>>(sBuf,  NF, 0.f);
        fill_kernel<<<cdiv(NF, 256), 256, 0, stream>>>(sqBuf, NF, 0.f);
        fill_kernel<<<cdiv(NF, 256), 256, 0, stream>>>(mnBuf, NF,  INFINITY);
        fill_kernel<<<cdiv(NF, 256), 256, 0, stream>>>(mxBuf, NF, -INFINITY);
        fill_kernel<<<cdiv(NN, 256), 256, 0, stream>>>(cnt, NN, 0.f);
        count_kernel<<<cdiv(NE, 256), 256, 0, stream>>>(tgtI, cnt, NE);
        aggregate_kernel<<<cdiv(EF, 256), 256, 0, stream>>>(h, tgtI, sBuf, sqBuf, mnBuf, mxBuf, NE, F);
        build_post_kernel<<<cdiv(NF, 256), 256, 0, stream>>>(xcur, sBuf, sqBuf, mnBuf, mxBuf, cnt, P, NN, F);

        // post_nn then lin (aggregate buffers recycled as outputs)
        gemm(P,    nullptr, NN, 13 * F, 13 * F, cv[l].postW, wpack, cv[l].postb, sBuf,  HD, HD, 0, 0, stream);
        gemm(sBuf, nullptr, NN, HD,     HD,     cv[l].linW,  wpack, cv[l].linb,  sqBuf, HD, HD, 0, 0, stream);

        // batch-norm (batch statistics) + relu
        fill_kernel<<<1, HD, 0, stream>>>(csum, HD, 0.f);
        fill_kernel<<<1, HD, 0, stream>>>(csq,  HD, 0.f);
        long long NH = (long long)NN * HD;
        colstats_kernel<<<cdiv(NH, 256), 256, 0, stream>>>(sqBuf, NN, HD, csum, csq);
        bn_relu_kernel<<<cdiv(NH, 256), 256, 0, stream>>>(sqBuf, csum, csq, cv[l].bn_g, cv[l].bn_b, xping, NN, HD);

        xcur = xping;
        float* t = xping; xping = xpong; xpong = t;
        F = HD;
    }

    // Set2Set (processing_steps=1, zero initial LSTM state)
    s2s_q_kernel<<<cdiv(BD * HD, 128), 128, 0, stream>>>(bih, bhh, q, BD, HD);
    s2s_e_kernel<<<cdiv(NN, 128), 128, 0, stream>>>(xcur, batch, q, ev, NN, HD);
    fill_kernel<<<1, BD, 0, stream>>>(mseg, BD, -INFINITY);
    s2s_max_kernel<<<cdiv(NN, 256), 256, 0, stream>>>(ev, batch, mseg, NN);
    fill_kernel<<<1, BD, 0, stream>>>(asum, BD, 0.f);
    s2s_a_kernel<<<cdiv(NN, 256), 256, 0, stream>>>(ev, batch, mseg, av, asum, NN);
    fill_kernel<<<cdiv(BD * HD, 128), 128, 0, stream>>>(rr, BD * HD, 0.f);
    s2s_r_kernel<<<cdiv((long long)NN * HD, 256), 256, 0, stream>>>(xcur, batch, av, asum, rr, NN, HD);
    s2s_qstar_kernel<<<cdiv(BD * 2 * HD, 128), 128, 0, stream>>>(q, rr, qstar, BD, HD);

    // head MLPs (B=10, trivial)
    mlp_kernel<<<cdiv(BD * HD, 128), 128, 0, stream>>>(qstar, BD, 2 * HD, lin1W, lin1b, t1, HD, 1);
    mlp_kernel<<<cdiv(BD * 64, 128), 128, 0, stream>>>(t1, BD, HD, lin2W, lin2b, t2, 64, 1);
    mlp_kernel<<<cdiv(BD * 2, 64), 64, 0, stream>>>(t2, BD, 64, lin3W, lin3b, (float*)d_out, 2, 0);
}
